// STEGOLOSSv2_48696339202758
// MI455X (gfx1250) — matrix operand order repaired and checked by reading, not compiled, not run on hardware
//
#include <hip/hip_runtime.h>
#include <hip/hip_bf16.h>
#include <math.h>

typedef _Float16 half8  __attribute__((ext_vector_type(8)));
typedef _Float16 v16h   __attribute__((ext_vector_type(16)));
typedef float    v8f    __attribute__((ext_vector_type(8)));

#define NNB   8      // batches
#define CF    384    // feat channels
#define CC    90     // code channels
#define CCP   96     // padded code channels
#define HH    56
#define WW    56
#define SS    32
#define POS   1024   // SS*SS
#define NPAIR 40     // 5 perms * 8 batches

// ---------------------------------------------------------------------------
// bilinear weights helper (border padding, align_corners=True, 56x56 source)
// ---------------------------------------------------------------------------
__device__ inline void bilin(float x, float y,
                             int& x0, int& x1, int& y0, int& y1,
                             float& w00, float& w01, float& w10, float& w11) {
    float gx = fminf(fmaxf((x + 1.0f) * 0.5f * (WW - 1), 0.0f), (float)(WW - 1));
    float gy = fminf(fmaxf((y + 1.0f) * 0.5f * (HH - 1), 0.0f), (float)(HH - 1));
    float fx = floorf(gx), fy = floorf(gy);
    float wx = gx - fx,   wy = gy - fy;
    x0 = (int)fx; y0 = (int)fy;
    x1 = min(x0 + 1, WW - 1);
    y1 = min(y0 + 1, HH - 1);
    w00 = (1.f - wx) * (1.f - wy);
    w01 = wx * (1.f - wy);
    w10 = (1.f - wx) * wy;
    w11 = wx * wy;
}

// job -> (image index, coord pointer, pair slot). jobs 0..7 = positive (c1),
// jobs 8..47 = negative pairs (perm i, batch n) sampled at c2.
__device__ inline void job_decode(int b, int pos, const float* coords1,
                                  const float* coords2, const int* perms,
                                  int& img, const float*& cptr, int& pp) {
    int h = pos >> 5, w = pos & 31;          // output row index = h*32+w
    if (b < NNB) {
        img = b; pp = -1;
        // transposed grid: c1[n,h,w] = coords1[n,w,h]
        cptr = coords1 + (((size_t)b * SS + w) * SS + h) * 2;
    } else {
        int q = b - NNB;
        int n = q & 7, i = q >> 3;
        img = perms[i * NNB + n];
        pp  = q;
        cptr = coords2 + (((size_t)n * SS + w) * SS + h) * 2;
    }
}

// ---------------------------------------------------------------------------
// Kernel 1: grid-sample feats + channel L2-normalize -> f16, K-contiguous rows
// one block (384 threads) per (job, position)
// ---------------------------------------------------------------------------
__global__ void sample_feats_kernel(const float* __restrict__ feats,
                                    const float* __restrict__ coords1,
                                    const float* __restrict__ coords2,
                                    const int*   __restrict__ perms,
                                    _Float16* __restrict__ F1,   // [8][1024][384]
                                    _Float16* __restrict__ F2) { // [40][1024][384]
    __shared__ float sh[512];
    int job = blockIdx.x;
    int pos = job & (POS - 1);
    int b   = job >> 10;
    int img, pp; const float* cp;
    job_decode(b, pos, coords1, coords2, perms, img, cp, pp);

    int x0, x1, y0, y1; float w00, w01, w10, w11;
    bilin(cp[0], cp[1], x0, x1, y0, y1, w00, w01, w10, w11);

    int c = threadIdx.x;                     // 0..383
    size_t base = ((size_t)img * CF + c) * HH;
    const float* r0 = feats + (base + y0) * WW;
    const float* r1 = feats + (base + y1) * WW;
    float v = r0[x0] * w00 + r0[x1] * w01 + r1[x0] * w10 + r1[x1] * w11;

    sh[c] = v * v;
    if (c < 128) sh[CF + c] = 0.0f;
    __syncthreads();
    for (int s = 256; s > 0; s >>= 1) {
        if (c < s) sh[c] += sh[c + s];
        __syncthreads();
    }
    float scale = 1.0f / fmaxf(sqrtf(sh[0]), 1e-10f);

    _Float16* dst = (pp < 0) ? (F1 + ((size_t)b * POS + pos) * CF)
                             : (F2 + ((size_t)pp * POS + pos) * CF);
    dst[c] = (_Float16)(v * scale);
}

// ---------------------------------------------------------------------------
// Kernel 2: grid-sample code + log_softmax -> lp (f16, pos only), q (f16),
// ent = sum(q*lq) (f32).  one block (128 threads) per (job, position)
// ---------------------------------------------------------------------------
__global__ void sample_code_kernel(const float* __restrict__ code,
                                   const float* __restrict__ coords1,
                                   const float* __restrict__ coords2,
                                   const int*   __restrict__ perms,
                                   _Float16* __restrict__ LP1,   // [8][1024][96]
                                   _Float16* __restrict__ Q1,    // [8][1024][96]
                                   float*    __restrict__ ENT1,  // [8][1024]
                                   _Float16* __restrict__ QN,    // [40][1024][96]
                                   float*    __restrict__ ENTN) {// [40][1024]
    __shared__ float sh[128];
    int job = blockIdx.x;
    int pos = job & (POS - 1);
    int b   = job >> 10;
    int img, pp; const float* cp;
    job_decode(b, pos, coords1, coords2, perms, img, cp, pp);

    int x0, x1, y0, y1; float w00, w01, w10, w11;
    bilin(cp[0], cp[1], x0, x1, y0, y1, w00, w01, w10, w11);

    int c = threadIdx.x;                     // 0..127, active for c<90
    bool act = (c < CC);
    float s = -3.0e38f;
    if (act) {
        size_t base = ((size_t)img * CC + c) * HH;
        const float* r0 = code + (base + y0) * WW;
        const float* r1 = code + (base + y1) * WW;
        s = r0[x0] * w00 + r0[x1] * w01 + r1[x0] * w10 + r1[x1] * w11;
    }
    sh[c] = s; __syncthreads();
    for (int st = 64; st > 0; st >>= 1) {
        if (c < st) sh[c] = fmaxf(sh[c], sh[c + st]);
        __syncthreads();
    }
    float mx = sh[0]; __syncthreads();

    float e = act ? expf(s - mx) : 0.0f;
    sh[c] = e; __syncthreads();
    for (int st = 64; st > 0; st >>= 1) {
        if (c < st) sh[c] += sh[c + st];
        __syncthreads();
    }
    float lse = logf(sh[0]); __syncthreads();

    float lp = s - mx - lse;                 // log_softmax
    float q  = act ? expf(lp) : 0.0f;
    sh[c] = act ? q * lp : 0.0f; __syncthreads();
    for (int st = 64; st > 0; st >>= 1) {
        if (c < st) sh[c] += sh[c + st];
        __syncthreads();
    }
    float ent = sh[0];

    if (pp < 0) {
        size_t o = ((size_t)b * POS + pos) * CCP;
        if (c < CCP) {
            LP1[o + c] = act ? (_Float16)lp : (_Float16)0.0f;
            Q1 [o + c] = act ? (_Float16)q  : (_Float16)0.0f;
        }
        if (c == 0) ENT1[(size_t)b * POS + pos] = ent;
    } else {
        size_t o = ((size_t)pp * POS + pos) * CCP;
        if (c < CCP) QN[o + c] = act ? (_Float16)q : (_Float16)0.0f;
        if (c == 0) ENTN[(size_t)pp * POS + pos] = ent;
    }
}

// ---------------------------------------------------------------------------
// WMMA fragment loaders (V_WMMA_*_16X16X32 f16 layouts, wave32)
// A (16x32): lane row M=lane&15; K-chunks at kb+[0..7] and kb+16+[0..7], kb=(lane&16)?8:0
// B (32x16): lane col N=lane&15; K-chunks at kb+[0..7] and kb+16+[0..7], kb=(lane&16)?8:0
// ---------------------------------------------------------------------------
__device__ inline v16h load_a_frag(const _Float16* row, int kk, int lane) {
    int kb = (lane & 16) ? 8 : 0;
    half8 c0 = *(const half8*)(row + kk + kb);
    half8 c1 = *(const half8*)(row + kk + kb + 16);
    v16h r;
#pragma unroll
    for (int i = 0; i < 8; ++i) { r[i] = c0[i]; r[i + 8] = c1[i]; }
    return r;
}
__device__ inline v16h load_b_frag(const _Float16* row, int kk, int lane) {
    int kb = (lane & 16) ? 8 : 0;
    half8 c0 = *(const half8*)(row + kk + kb);
    half8 c1 = *(const half8*)(row + kk + kb + 16);
    v16h r;
#pragma unroll
    for (int i = 0; i < 8; ++i) { r[i] = c0[i]; r[i + 8] = c1[i]; }
    return r;
}

__device__ inline v8f wmma_step(v16h a, v16h b, v8f c) {
    return __builtin_amdgcn_wmma_f32_16x16x32_f16(
        false, a, false, b, (short)0, c, false, false);
}

// ---------------------------------------------------------------------------
// Kernel 3: each wave computes a 32x32 macro-tile (2x2 of 16x16 WMMA tiles)
// with register blocking: per k-step load 2 A-frags + 2 B-frags, issue 4 WMMAs
// (halves L2 fragment traffic vs 1 tile/wave, and the 4 independent WMMAs
// cover the f16-WMMA RAW latency). fd: K=384 (48 WMMAs), cross: K=96 (12).
// Then pointwise tanh loss over 32 outputs/lane, wave-reduce, 1 atomic.
// 256 threads = 8 waves, one macro-tile per wave.
// ---------------------------------------------------------------------------
__global__ void loss_kernel(const _Float16* __restrict__ FA,
                            const _Float16* __restrict__ FB,
                            const _Float16* __restrict__ LPA,
                            const _Float16* __restrict__ QB,
                            const float*    __restrict__ ENTB,
                            float* __restrict__ out_slot,
                            int neg, float scale) {
    int lane = threadIdx.x & 31;
    int wid  = blockIdx.x * 8 + (threadIdx.x >> 5);
    int b    = wid >> 10;                    // macro-tiles per batch = 32*32
    int t    = wid & 1023;
    int m0   = (t >> 5) << 5;                // hw macro-tile origin (32 rows)
    int n0   = (t & 31) << 5;                // ij macro-tile origin (32 cols)
    int ba   = neg ? (b & 7) : b;            // A side always from batch n
    int bb   = b;                            // B side: batch or pair slot

    const _Float16* Arow0 = FA + ((size_t)ba * POS + m0 +      (lane & 15)) * CF;
    const _Float16* Arow1 = FA + ((size_t)ba * POS + m0 + 16 + (lane & 15)) * CF;
    const _Float16* Brow0 = FB + ((size_t)bb * POS + n0 +      (lane & 15)) * CF;
    const _Float16* Brow1 = FB + ((size_t)bb * POS + n0 + 16 + (lane & 15)) * CF;

    v8f fd00 = {}, fd01 = {}, fd10 = {}, fd11 = {};
#pragma unroll
    for (int kk = 0; kk < CF; kk += 32) {
        v16h a0 = load_a_frag(Arow0, kk, lane);
        v16h a1 = load_a_frag(Arow1, kk, lane);
        v16h b0 = load_b_frag(Brow0, kk, lane);
        v16h b1 = load_b_frag(Brow1, kk, lane);
        fd00 = wmma_step(a0, b0, fd00);
        fd01 = wmma_step(a0, b1, fd01);
        fd10 = wmma_step(a1, b0, fd10);
        fd11 = wmma_step(a1, b1, fd11);
    }

    const _Float16* LProw0 = LPA + ((size_t)ba * POS + m0 +      (lane & 15)) * CCP;
    const _Float16* LProw1 = LPA + ((size_t)ba * POS + m0 + 16 + (lane & 15)) * CCP;
    const _Float16* Qrow0  = QB  + ((size_t)bb * POS + n0 +      (lane & 15)) * CCP;
    const _Float16* Qrow1  = QB  + ((size_t)bb * POS + n0 + 16 + (lane & 15)) * CCP;

    v8f cr00 = {}, cr01 = {}, cr10 = {}, cr11 = {};
#pragma unroll
    for (int kk = 0; kk < CCP; kk += 32) {
        v16h a0 = load_a_frag(LProw0, kk, lane);
        v16h a1 = load_a_frag(LProw1, kk, lane);
        v16h b0 = load_b_frag(Qrow0, kk, lane);
        v16h b1 = load_b_frag(Qrow1, kk, lane);
        cr00 = wmma_step(a0, b0, cr00);
        cr01 = wmma_step(a0, b1, cr01);
        cr10 = wmma_step(a1, b0, cr10);
        cr11 = wmma_step(a1, b1, cr11);
    }

    // ent indexed by output column: col = n0 + 16*nj + (lane&15)
    float ent0 = ENTB[(size_t)bb * POS + n0 +      (lane & 15)];
    float ent1 = ENTB[(size_t)bb * POS + n0 + 16 + (lane & 15)];

    float s = 0.0f;
#pragma unroll
    for (int r = 0; r < 8; ++r) {
        float d;
        d = fd00[r] + tanhf(0.5f * (ent0 - cr00[r] - 1.1f)); s += d * d;
        d = fd01[r] + tanhf(0.5f * (ent1 - cr01[r] - 1.1f)); s += d * d;
        d = fd10[r] + tanhf(0.5f * (ent0 - cr10[r] - 1.1f)); s += d * d;
        d = fd11[r] + tanhf(0.5f * (ent1 - cr11[r] - 1.1f)); s += d * d;
    }
#pragma unroll
    for (int off = 16; off > 0; off >>= 1)
        s += __shfl_xor(s, off, 32);
    if (lane == 0) atomicAdd(out_slot, s * scale);
}

__global__ void zero_out_kernel(float* out) {
    if (threadIdx.x < 2) out[threadIdx.x] = 0.0f;
}

// ---------------------------------------------------------------------------
extern "C" void kernel_launch(void* const* d_in, const int* in_sizes, int n_in,
                              void* d_out, int out_size, void* d_ws, size_t ws_size,
                              hipStream_t stream) {
    const float* feats   = (const float*)d_in[0];
    const float* code    = (const float*)d_in[1];
    const float* coords1 = (const float*)d_in[2];
    const float* coords2 = (const float*)d_in[3];
    const int*   perms   = (const int*)d_in[4];
    float* out = (float*)d_out;

    // workspace partition
    char* p = (char*)d_ws;
    _Float16* F1  = (_Float16*)p; p += (size_t)NNB  * POS * CF  * 2;
    _Float16* F2  = (_Float16*)p; p += (size_t)NPAIR* POS * CF  * 2;
    _Float16* LP1 = (_Float16*)p; p += (size_t)NNB  * POS * CCP * 2;
    _Float16* Q1  = (_Float16*)p; p += (size_t)NNB  * POS * CCP * 2;
    _Float16* QN  = (_Float16*)p; p += (size_t)NPAIR* POS * CCP * 2;
    float*    ENT1= (float*)p;    p += (size_t)NNB  * POS * 4;
    float*    ENTN= (float*)p;    p += (size_t)NPAIR* POS * 4;

    zero_out_kernel<<<1, 32, 0, stream>>>(out);

    int jobs = (NNB + NPAIR) * POS;          // 48 * 1024
    sample_feats_kernel<<<jobs, CF, 0, stream>>>(feats, coords1, coords2, perms, F1, F2);
    sample_code_kernel<<<jobs, 128, 0, stream>>>(code, coords1, coords2, perms,
                                                 LP1, Q1, ENT1, QN, ENTN);

    // positive: 8 batches * 1024 macro-tiles, 8 waves/block
    float pos_scale = 1.0f / (float)((size_t)NNB * POS * POS);
    loss_kernel<<<(NNB * 1024) / 8, 256, 0, stream>>>(
        F1, F1, LP1, Q1, ENT1, out + 0, /*neg=*/0, pos_scale);

    // negatives: 40 pair-slots * 1024 macro-tiles
    float neg_scale = 1.0f / (float)((size_t)NPAIR * POS * POS);
    loss_kernel<<<(NPAIR * 1024) / 8, 256, 0, stream>>>(
        F1, F2, LP1, QN, ENTN, out + 1, /*neg=*/1, neg_scale);
}
